// InputLayer_16999480557932
// MI455X (gfx1250) — compile-verified, matrix operand order checked by reading, not run on hardware
//
#include <hip/hip_runtime.h>

typedef __attribute__((ext_vector_type(16))) _Float16 v16h;
typedef __attribute__((ext_vector_type(8)))  float    v8f;

#define D_MODEL 512
#define NFEAT   32

// dt(d) = exp(-2d * ln(10000)/512) = exp2(d * -log2(10000)/256)
#define NEG_LOG2_1E4_OVER_256 (-0.051905126482615036f)

__global__ __launch_bounds__(256) void sincos_embed_wmma(
    const float* __restrict__ x,   // [B, 32]
    float* __restrict__ out)       // [B, 512]
{
  const int lane = threadIdx.x & 31;
  const int wave = threadIdx.x >> 5;
  const int b    = blockIdx.x * 8 + wave;          // one batch row per wave

  const int m     = lane & 15;                     // A-matrix row this lane owns
  const int kbase = (lane < 16) ? 0 : 16;          // which half of the feature row

  // Cache this lane's 16 features in registers; reused across all 16 d-tiles.
  const float* xrow = x + b * NFEAT + kbase;       // 16B-aligned
  float xv[16];
#pragma unroll
  for (int i = 0; i < 4; ++i) {
    float4 t = ((const float4*)xrow)[i];
    xv[4*i+0] = t.x; xv[4*i+1] = t.y; xv[4*i+2] = t.z; xv[4*i+3] = t.w;
  }

  // B matrix: constant 1/32 -> D = mean over K (feature axis). 1/32 exact in f16.
  v16h bmat;
#pragma unroll
  for (int i = 0; i < 16; ++i) bmat[i] = (_Float16)0.03125f;

  const v8f zero = {};

#pragma unroll 1
  for (int t = 0; t < 16; ++t) {                   // 16 tiles of 16 freqs = 256
    const int   d  = t * 16 + m;
    const float dt = __builtin_amdgcn_exp2f((float)d * NEG_LOG2_1E4_OVER_256);

    v16h a_sin, a_cos;
#pragma unroll
    for (int k = 0; k < 16; ++k) {
      float si = xv[k] * dt;
      a_sin[k] = (_Float16)__sinf(si);
      a_cos[k] = (_Float16)__cosf(si);
    }

    // Feature-mean via matrix pipe: D[m, n] = sum_k A[m,k] * (1/32), all n equal.
    v8f dsin = __builtin_amdgcn_wmma_f32_16x16x32_f16(
        false, a_sin, false, bmat, (short)0, zero, false, false);
    v8f dcos = __builtin_amdgcn_wmma_f32_16x16x32_f16(
        false, a_cos, false, bmat, (short)0, zero, false, false);

    // Lane L emits output column j = 32*t + L = 2*d + (sin:0 / cos:1),
    // value lives in D-VGPR r = (L>>1)&7 of this lane. Select via cndmask chain.
    const int r = (lane >> 1) & 7;
    float vs = dsin[0], vc = dcos[0];
#pragma unroll
    for (int rr = 1; rr < 8; ++rr) {
      if (r == rr) { vs = dsin[rr]; vc = dcos[rr]; }
    }
    float val = (lane & 1) ? vc : vs;
    out[b * D_MODEL + t * 32 + lane] = val;        // coalesced 128B per wave
  }
}

extern "C" void kernel_launch(void* const* d_in, const int* in_sizes, int n_in,
                              void* d_out, int out_size, void* d_ws, size_t ws_size,
                              hipStream_t stream) {
  const float* x  = (const float*)d_in[0];
  float* out      = (float*)d_out;
  const int batch = in_sizes[0] / NFEAT;           // 16384
  const int blocks = batch / 8;                    // 8 waves (rows) per 256-thread block
  sincos_embed_wmma<<<blocks, 256, 0, stream>>>(x, out);
}
